// TrXLI_1511828488255
// MI455X (gfx1250) — matrix operand order repaired and checked by reading, not compiled, not run on hardware
//
#include <hip/hip_runtime.h>
#include <hip/hip_bf16.h>

// ---------------------------------------------------------------------------
// Transformer block on gfx1250 (wave32, WMMA f16 -> f32 accumulate)
// B=4, S=2048, D=512, H=8, dh=64
// Round 2: register-double-buffered GEMMs (32x64 per wave) + async-to-LDS
// double-buffered K/V staging in attention (ASYNCcnt pipeline).
// ---------------------------------------------------------------------------

#define BATCH 4
#define SEQ   2048
#define DIM   512
#define NH    8
#define DH    64
#define BS    (BATCH * SEQ)   // 8192 rows

typedef __attribute__((ext_vector_type(16))) _Float16 v16h;
typedef __attribute__((ext_vector_type(8)))  _Float16 v8h;
typedef __attribute__((ext_vector_type(8)))  float    v8f;

// ---- WMMA fragment helpers (layouts per CDNA5 ISA 7.12.2) ------------------

// A fragment: 16x32 f16, row-major source, stride lda (elements).
// lane 0-15: row=lane, K = 0..7 and 16..23 ; lane 16-31: row=lane-16, K = 8..15 and 24..31
__device__ __forceinline__ v16h load_frag_a(const _Float16* p, int lda, int lane) {
  int row  = lane & 15;
  int koff = (lane & 16) ? 8 : 0;
  const _Float16* r = p + (size_t)row * lda + koff;
  v8h lo = *(const v8h*)r;          // K = koff .. koff+7
  v8h hi = *(const v8h*)(r + 16);   // K = koff+16 .. koff+23
  v16h a;
#pragma unroll
  for (int i = 0; i < 8; ++i) { a[i] = lo[i]; a[8 + i] = hi[i]; }
  return a;
}

// B fragment: 32x16 f16, K x N, from storage where each COLUMN n is contiguous
// over K with stride ldb between columns (i.e. B^T row-major).
// lane 0-15: col=lane, K = 0..7 and 16..23 ; lane 16-31: col=lane-16, K = 8..15 and 24..31
__device__ __forceinline__ v16h load_frag_b(const _Float16* bt, int ldb, int lane) {
  int n  = lane & 15;
  int kb = (lane & 16) ? 8 : 0;
  const _Float16* c = bt + (size_t)n * ldb + kb;
  v8h lo = *(const v8h*)c;
  v8h hi = *(const v8h*)(c + 16);
  v16h b;
#pragma unroll
  for (int i = 0; i < 8; ++i) { b[i] = lo[i]; b[8 + i] = hi[i]; }
  return b;
}

__device__ __forceinline__ v8f wmma_f16(v16h a, v16h b, v8f c) {
  return __builtin_amdgcn_wmma_f32_16x16x32_f16(false, a, false, b, (short)0, c,
                                                false, false);
}

// Async copy 16B global -> LDS, tracked by ASYNCcnt (gfx1250).
// dsaddr comes from the low 32 bits of the generic LDS pointer (aperture keeps
// the LDS byte address in addr[31:0] per the flat-addressing rules).
__device__ __forceinline__ void async_b128(const _Float16* lds_dst,
                                           const _Float16* gsrc) {
  asm volatile("global_load_async_to_lds_b128 %0, %1, off"
               :: "v"((unsigned)(size_t)lds_dst), "v"(gsrc)
               : "memory");
}

// ---- Prep: convert & transpose weights to f16 -----------------------------
// W [H][K][N] fp32 row-major  ->  WT [H][N][K] f16 row-major
__global__ void cvt_transpose_h(const float* __restrict__ W,
                                _Float16* __restrict__ WT,
                                int H, int K, int N) {
  int idx = blockIdx.x * blockDim.x + threadIdx.x;
  if (idx >= H * K * N) return;
  int n = idx % N;
  int k = (idx / N) % K;
  int h = idx / (N * K);
  WT[((size_t)h * N + n) * K + k] = (_Float16)W[idx];
}

// ---- LayerNorm (row of 512) -> f16 ----------------------------------------
__global__ __launch_bounds__(128) void layernorm_f16(const float* __restrict__ X,
                                                     const float* __restrict__ g,
                                                     const float* __restrict__ bb,
                                                     _Float16* __restrict__ Y) {
  __shared__ float red[256];
  int row = blockIdx.x;
  const float* x = X + (size_t)row * DIM;
  float s = 0.f, ss = 0.f;
  for (int i = threadIdx.x; i < DIM; i += 128) {
    float v = x[i];
    s += v; ss += v * v;
  }
  red[threadIdx.x] = s;
  red[threadIdx.x + 128] = ss;
  __syncthreads();
  for (int off = 64; off > 0; off >>= 1) {
    if ((int)threadIdx.x < off) {
      red[threadIdx.x]       += red[threadIdx.x + off];
      red[threadIdx.x + 128] += red[threadIdx.x + 128 + off];
    }
    __syncthreads();
  }
  float mean = red[0] * (1.0f / DIM);
  float var  = red[128] * (1.0f / DIM) - mean * mean;
  float rstd = rsqrtf(var + 1e-5f);
  for (int i = threadIdx.x; i < DIM; i += 128) {
    float v = (x[i] - mean) * rstd * g[i] + bb[i];
    Y[(size_t)row * DIM + i] = (_Float16)v;
  }
}

// ---- QKV projection --------------------------------------------------------
// grid (S/256, H, B*3); block 256 (8 waves); wave -> 32 rows x 64 cols,
// register double-buffered over K.
// q,k row-major [B,H,S,64] ; v stored transposed [B,H,64,S]
__global__ __launch_bounds__(256) void qkv_gemm(
    const _Float16* __restrict__ hF16,
    const _Float16* __restrict__ WqT, const _Float16* __restrict__ WkT,
    const _Float16* __restrict__ WvT,
    const float* __restrict__ bq, const float* __restrict__ bk,
    const float* __restrict__ bv,
    _Float16* __restrict__ q, _Float16* __restrict__ k,
    _Float16* __restrict__ vT) {
  const int lane = threadIdx.x & 31;
  const int wave = threadIdx.x >> 5;
  const int srow = blockIdx.x * 256 + wave * 32;
  const int h    = blockIdx.y;
  const int b    = blockIdx.z / 3;
  const int mat  = blockIdx.z % 3;

  const _Float16* WT   = (mat == 0) ? WqT : (mat == 1) ? WkT : WvT;
  const float*    bias = (mat == 0) ? bq  : (mat == 1) ? bk  : bv;
  const _Float16* A  = hF16 + ((size_t)b * SEQ + srow) * DIM;
  const _Float16* Bh = WT + (size_t)h * DH * DIM;   // [64][512] f16

  v8f acc[2][4] = {};
  v16h a0 = load_frag_a(A, DIM, lane);
  v16h a1 = load_frag_a(A + (size_t)16 * DIM, DIM, lane);
  v16h bf0 = load_frag_b(Bh + (size_t)0  * DIM, DIM, lane);
  v16h bf1 = load_frag_b(Bh + (size_t)16 * DIM, DIM, lane);
  v16h bf2 = load_frag_b(Bh + (size_t)32 * DIM, DIM, lane);
  v16h bf3 = load_frag_b(Bh + (size_t)48 * DIM, DIM, lane);

  for (int kk = 0; kk < DIM; kk += 32) {
    int nk = kk + 32;
    v16h na0, na1, nb0, nb1, nb2, nb3;
    if (nk < DIM) {   // prefetch next K-step fragments
      na0 = load_frag_a(A + nk, DIM, lane);
      na1 = load_frag_a(A + (size_t)16 * DIM + nk, DIM, lane);
      nb0 = load_frag_b(Bh + (size_t)0  * DIM + nk, DIM, lane);
      nb1 = load_frag_b(Bh + (size_t)16 * DIM + nk, DIM, lane);
      nb2 = load_frag_b(Bh + (size_t)32 * DIM + nk, DIM, lane);
      nb3 = load_frag_b(Bh + (size_t)48 * DIM + nk, DIM, lane);
    }
    acc[0][0] = wmma_f16(a0, bf0, acc[0][0]);
    acc[0][1] = wmma_f16(a0, bf1, acc[0][1]);
    acc[0][2] = wmma_f16(a0, bf2, acc[0][2]);
    acc[0][3] = wmma_f16(a0, bf3, acc[0][3]);
    acc[1][0] = wmma_f16(a1, bf0, acc[1][0]);
    acc[1][1] = wmma_f16(a1, bf1, acc[1][1]);
    acc[1][2] = wmma_f16(a1, bf2, acc[1][2]);
    acc[1][3] = wmma_f16(a1, bf3, acc[1][3]);
    if (nk < DIM) {
      a0 = na0; a1 = na1;
      bf0 = nb0; bf1 = nb1; bf2 = nb2; bf3 = nb3;
    }
  }

  const int n0    = lane & 15;
  const int mbase = (lane >> 4) << 3;
  const size_t headoff = (size_t)b * NH + h;
#pragma unroll
  for (int sub = 0; sub < 2; ++sub) {
#pragma unroll
    for (int t = 0; t < 4; ++t) {
      int n = t * 16 + n0;
      float bia = bias[h * DH + n];
#pragma unroll
      for (int i = 0; i < 8; ++i) {
        float val = acc[sub][t][i] + bia;
        int m = srow + sub * 16 + mbase + i;
        if (mat == 2) {
          vT[(headoff * DH + n) * SEQ + m] = (_Float16)val;
        } else {
          _Float16* dst = (mat == 0) ? q : k;
          dst[(headoff * SEQ + m) * DH + n] = (_Float16)val;
        }
      }
    }
  }
}

// ---- Flash attention with async-to-LDS double-buffered K/V ----------------
// grid (S/128, H, B); block 256 (8 waves); wave -> 16 queries x 64 dims.
// All 8 waves share the K/V tile of 32 keys staged in LDS via
// global_load_async_to_lds_b128 (ASYNCcnt), double-buffered.
__global__ __launch_bounds__(256) void attn_kernel(
    const _Float16* __restrict__ q, const _Float16* __restrict__ k,
    const _Float16* __restrict__ vT, _Float16* __restrict__ om) {
  __shared__ __align__(16) _Float16 Kt[2][32 * DH];   // [keys][dh]     2x4KB
  __shared__ __align__(16) _Float16 Vt[2][DH * 32];   // [dh][keys]     2x4KB
  __shared__ __align__(16) _Float16 plds[8][16 * 32]; // per-wave P     8KB

  const int tid  = threadIdx.x;
  const int lane = tid & 31;
  const int wave = tid >> 5;
  const int h = blockIdx.y, b = blockIdx.z;
  const int qrow = blockIdx.x * 128 + wave * 16;
  const size_t headoff = (size_t)b * NH + h;
  const _Float16* qh = q  + headoff * SEQ * DH;
  const _Float16* kh = k  + headoff * SEQ * DH;
  const _Float16* vh = vT + headoff * DH * SEQ;   // [64][S]

  const int n0    = lane & 15;
  const int mbase = (lane >> 4) << 3;

  // cooperative copy mapping: each thread moves 16B of K and 16B of V per tile
  const int krow = tid >> 3, kch = (tid & 7) * 8;   // K: 32 rows x 128B
  const int vrow = tid >> 2, vch = (tid & 3) * 8;   // V: 64 rows x 64B

  // prologue: stage tile 0
  async_b128(&Kt[0][krow * DH + kch], kh + (size_t)krow * DH + kch);
  async_b128(&Vt[0][vrow * 32 + vch], vh + (size_t)vrow * SEQ + vch);

  v16h qa0 = load_frag_a(qh + (size_t)qrow * DH,      DH, lane);
  v16h qa1 = load_frag_a(qh + (size_t)qrow * DH + 32, DH, lane);

  v8f oacc[4] = {};
  float mrow[8], lrow[8];
#pragma unroll
  for (int i = 0; i < 8; ++i) { mrow[i] = -1e30f; lrow[i] = 0.f; }
  const float scale = 0.125f;   // 1/sqrt(64)
  _Float16* pl = plds[wave];

  const int NTILE = SEQ / 32;   // 64
  for (int it = 0; it < NTILE; ++it) {
    const int kt  = it * 32;
    const int buf = it & 1;
    if (it + 1 < NTILE) {
      // prefetch next tile into the other buffer (read-complete via trailing
      // barrier of the previous iteration)
      const int nkt = kt + 32;
      async_b128(&Kt[buf ^ 1][krow * DH + kch], kh + (size_t)(nkt + krow) * DH + kch);
      async_b128(&Vt[buf ^ 1][vrow * 32 + vch], vh + (size_t)vrow * SEQ + nkt + vch);
      asm volatile("s_wait_asynccnt 0x2" ::: "memory");  // current tile landed
    } else {
      asm volatile("s_wait_asynccnt 0x0" ::: "memory");
    }
    __syncthreads();   // every wave's slice of the current tile is visible

    const _Float16* Kb = Kt[buf];
    const _Float16* Vb = Vt[buf];

    // scores: 16 queries x 32 keys (two 16-key N tiles, K=64 -> 2 WMMAs each)
    v8f s0 = {}, s1 = {};
    {
      v16h b0 = load_frag_b(Kb,                DH, lane);
      v16h b1 = load_frag_b(Kb + 32,           DH, lane);
      s0 = wmma_f16(qa0, b0, s0);
      s0 = wmma_f16(qa1, b1, s0);
      v16h b2 = load_frag_b(Kb + 16 * DH,      DH, lane);
      v16h b3 = load_frag_b(Kb + 16 * DH + 32, DH, lane);
      s1 = wmma_f16(qa0, b2, s1);
      s1 = wmma_f16(qa1, b3, s1);
    }
    float p0[8], p1[8];
#pragma unroll
    for (int i = 0; i < 8; ++i) {
      float a0 = s0[i] * scale, a1 = s1[i] * scale;
      float vmax = fmaxf(a0, a1);
#pragma unroll
      for (int off = 1; off < 16; off <<= 1)
        vmax = fmaxf(vmax, __shfl_xor(vmax, off, 32));
      float mnew  = fmaxf(mrow[i], vmax);
      float alpha = __expf(mrow[i] - mnew);
      mrow[i] = mnew;
      float e0 = __expf(a0 - mnew);
      float e1 = __expf(a1 - mnew);
      p0[i] = e0; p1[i] = e1;
      float rs = e0 + e1;
#pragma unroll
      for (int off = 1; off < 16; off <<= 1)
        rs += __shfl_xor(rs, off, 32);
      lrow[i] = lrow[i] * alpha + rs;
#pragma unroll
      for (int t = 0; t < 4; ++t) oacc[t][i] *= alpha;
    }
    // D-layout -> LDS -> A-layout relayout of P (f16); same-wave roundtrip
#pragma unroll
    for (int i = 0; i < 8; ++i) {
      pl[(mbase + i) * 32 + n0]      = (_Float16)p0[i];
      pl[(mbase + i) * 32 + 16 + n0] = (_Float16)p1[i];
    }
    asm volatile("s_wait_dscnt 0x0" ::: "memory");
    v16h pa = load_frag_a(pl, 32, lane);
#pragma unroll
    for (int t = 0; t < 4; ++t) {
      v16h vb = load_frag_b(Vb + (size_t)(t * 16) * 32, 32, lane);
      oacc[t] = wmma_f16(pa, vb, oacc[t]);
    }
    __syncthreads();   // all waves done reading buf before it is refilled
  }

  // normalize and store concat-head layout [B,S,512] (col = h*64 + e)
#pragma unroll
  for (int t = 0; t < 4; ++t) {
    int n = h * DH + t * 16 + n0;
#pragma unroll
    for (int i = 0; i < 8; ++i) {
      float val = oacc[t][i] / lrow[i];
      om[((size_t)b * SEQ + qrow + mbase + i) * DIM + n] = (_Float16)val;
    }
  }
}

// ---- 512x512 GEMM + bias + (relu) + residual, fp32 out --------------------
// grid (M/256, 512/64); block 256 (8 waves); wave -> 32 rows x 64 cols,
// register double-buffered over K.
__global__ __launch_bounds__(256) void gemm512_epilogue(
    const _Float16* __restrict__ A, const _Float16* __restrict__ WT,
    const float* __restrict__ bias, const float* __restrict__ res,
    float* __restrict__ out, int relu_mode) {
  const int lane = threadIdx.x & 31;
  const int wave = threadIdx.x >> 5;
  const int m0  = blockIdx.x * 256 + wave * 32;
  const int n0t = blockIdx.y * 64;

  const _Float16* Ar = A + (size_t)m0 * DIM;
  const _Float16* Wb = WT + (size_t)n0t * DIM;

  v8f acc[2][4] = {};
  v16h a0 = load_frag_a(Ar, DIM, lane);
  v16h a1 = load_frag_a(Ar + (size_t)16 * DIM, DIM, lane);
  v16h bf0 = load_frag_b(Wb + (size_t)0  * DIM, DIM, lane);
  v16h bf1 = load_frag_b(Wb + (size_t)16 * DIM, DIM, lane);
  v16h bf2 = load_frag_b(Wb + (size_t)32 * DIM, DIM, lane);
  v16h bf3 = load_frag_b(Wb + (size_t)48 * DIM, DIM, lane);

  for (int kk = 0; kk < DIM; kk += 32) {
    int nk = kk + 32;
    v16h na0, na1, nb0, nb1, nb2, nb3;
    if (nk < DIM) {
      na0 = load_frag_a(Ar + nk, DIM, lane);
      na1 = load_frag_a(Ar + (size_t)16 * DIM + nk, DIM, lane);
      nb0 = load_frag_b(Wb + (size_t)0  * DIM + nk, DIM, lane);
      nb1 = load_frag_b(Wb + (size_t)16 * DIM + nk, DIM, lane);
      nb2 = load_frag_b(Wb + (size_t)32 * DIM + nk, DIM, lane);
      nb3 = load_frag_b(Wb + (size_t)48 * DIM + nk, DIM, lane);
    }
    acc[0][0] = wmma_f16(a0, bf0, acc[0][0]);
    acc[0][1] = wmma_f16(a0, bf1, acc[0][1]);
    acc[0][2] = wmma_f16(a0, bf2, acc[0][2]);
    acc[0][3] = wmma_f16(a0, bf3, acc[0][3]);
    acc[1][0] = wmma_f16(a1, bf0, acc[1][0]);
    acc[1][1] = wmma_f16(a1, bf1, acc[1][1]);
    acc[1][2] = wmma_f16(a1, bf2, acc[1][2]);
    acc[1][3] = wmma_f16(a1, bf3, acc[1][3]);
    if (nk < DIM) {
      a0 = na0; a1 = na1;
      bf0 = nb0; bf1 = nb1; bf2 = nb2; bf3 = nb3;
    }
  }

  const int nl    = lane & 15;
  const int mbase = (lane >> 4) << 3;
#pragma unroll
  for (int sub = 0; sub < 2; ++sub) {
#pragma unroll
    for (int t = 0; t < 4; ++t) {
      int n = n0t + t * 16 + nl;
      float bia = bias[n];
#pragma unroll
      for (int i = 0; i < 8; ++i) {
        size_t idx = (size_t)(m0 + sub * 16 + mbase + i) * DIM + n;
        float v = acc[sub][t][i] + bia;
        if (relu_mode) v = fmaxf(v, 0.f);
        v += res[idx];
        out[idx] = v;
      }
    }
  }
}

// ---------------------------------------------------------------------------
extern "C" void kernel_launch(void* const* d_in, const int* in_sizes, int n_in,
                              void* d_out, int out_size, void* d_ws, size_t ws_size,
                              hipStream_t stream) {
  const float* x  = (const float*)d_in[0];
  const float* g1 = (const float*)d_in[1];
  const float* b1 = (const float*)d_in[2];
  const float* Wq = (const float*)d_in[3];
  const float* bq = (const float*)d_in[4];
  const float* Wk = (const float*)d_in[5];
  const float* bk = (const float*)d_in[6];
  const float* Wv = (const float*)d_in[7];
  const float* bv = (const float*)d_in[8];
  const float* Wm = (const float*)d_in[9];
  const float* bm = (const float*)d_in[10];
  const float* g2 = (const float*)d_in[11];
  const float* b2 = (const float*)d_in[12];
  const float* Wo = (const float*)d_in[13];
  const float* bo = (const float*)d_in[14];
  float* out = (float*)d_out;

  char* ws = (char*)d_ws;
  size_t off = 0;
  auto alloc = [&](size_t bytes) {
    void* p = ws + off;
    off = (off + bytes + 255) & ~(size_t)255;
    return p;
  };

  _Float16* hF16 = (_Float16*)alloc((size_t)BS * DIM * 2);        // LN1(x) f16
  _Float16* WqT  = (_Float16*)alloc((size_t)NH * DH * DIM * 2);
  _Float16* WkT  = (_Float16*)alloc((size_t)NH * DH * DIM * 2);
  _Float16* WvT  = (_Float16*)alloc((size_t)NH * DH * DIM * 2);
  _Float16* WmT  = (_Float16*)alloc((size_t)DIM * DIM * 2);
  _Float16* WoT  = (_Float16*)alloc((size_t)DIM * DIM * 2);
  _Float16* qb   = (_Float16*)alloc((size_t)BATCH * NH * SEQ * DH * 2);
  _Float16* kb   = (_Float16*)alloc((size_t)BATCH * NH * SEQ * DH * 2);
  _Float16* vTb  = (_Float16*)alloc((size_t)BATCH * NH * DH * SEQ * 2);
  _Float16* om   = (_Float16*)alloc((size_t)BS * DIM * 2);        // concat heads f16
  float*    hlp  = (float*)alloc((size_t)BS * DIM * 4);           // h_ln_post fp32
  _Float16* hfc  = (_Float16*)alloc((size_t)BS * DIM * 2);        // LN2 f16

  // 1) weight conversion / transpose
  {
    int tot = NH * DIM * DH;   // 262144
    cvt_transpose_h<<<(tot + 255) / 256, 256, 0, stream>>>(Wq, WqT, NH, DIM, DH);
    cvt_transpose_h<<<(tot + 255) / 256, 256, 0, stream>>>(Wk, WkT, NH, DIM, DH);
    cvt_transpose_h<<<(tot + 255) / 256, 256, 0, stream>>>(Wv, WvT, NH, DIM, DH);
    int tot2 = DIM * DIM;
    cvt_transpose_h<<<(tot2 + 255) / 256, 256, 0, stream>>>(Wm, WmT, 1, DIM, DIM);
    cvt_transpose_h<<<(tot2 + 255) / 256, 256, 0, stream>>>(Wo, WoT, 1, DIM, DIM);
  }

  // 2) LN1
  layernorm_f16<<<BS, 128, 0, stream>>>(x, g1, b1, hF16);

  // 3) QKV projections
  qkv_gemm<<<dim3(SEQ / 256, NH, BATCH * 3), 256, 0, stream>>>(
      hF16, WqT, WkT, WvT, bq, bk, bv, qb, kb, vTb);

  // 4) attention -> concat heads (f16)
  attn_kernel<<<dim3(SEQ / 128, NH, BATCH), 256, 0, stream>>>(qb, kb, vTb, om);

  // 5) h_ln_post = om @ Wm + bm + x   (fp32)
  gemm512_epilogue<<<dim3(BS / 256, DIM / 64), 256, 0, stream>>>(
      om, WmT, bm, x, hlp, 0);

  // 6) LN2
  layernorm_f16<<<BS, 128, 0, stream>>>(hlp, g2, b2, hfc);

  // 7) out = relu(hfc @ Wo + bo) + h_ln_post
  gemm512_epilogue<<<dim3(BS / 256, DIM / 64), 256, 0, stream>>>(
      hfc, WoT, bo, hlp, out, 1);
}
